// MLPNetwork_38869454029009
// MI455X (gfx1250) — matrix operand order repaired and checked by reading, not compiled, not run on hardware
//
#include <hip/hip_runtime.h>

// ---------------------------------------------------------------------------
// SLAYER SRM-alpha SNN forward:  x --W1--> psp -> spike --W2--> psp -> spike
// fc1 GEMM dominates (8e10 FLOP): f16 WMMA, 2x2 register blocking, split-K=4,
// LDS-transposed B staging so B fragments are two ds_load_b128 per tile.
// psp/spike recurrence streams u into LDS via async global->LDS (ASYNCcnt).
// ---------------------------------------------------------------------------

#define B_SZ   8
#define T_LEN  300
#define K_IN   32768          // 2*128*128
#define N_HID  512
#define N_OUT  19
#define THETA  10.0f
#define TAU_SR 10.0f
#define KTAPS  64
#define KSPLIT 4
#define KCH    (K_IN / KSPLIT) // 8192
#define XPITCH 40              // fc1 LDS row pitch in halves (conflict-free b128)
#define HSLOTS 128             // psp history ring (power of 2, >= 64 + prefetch)
#define HPITCH 132             // floats; multiple of 4 -> 16B-aligned b128 dests
#define APITCH 65              // refractory ring pitch (conflict-free)
#define PSP_BLK 64

#if defined(__has_builtin)
#if __has_builtin(__builtin_amdgcn_global_load_async_to_lds_b128) && \
    __has_builtin(__builtin_amdgcn_s_wait_asynccnt)
#define HAVE_ASYNC_LDS 1
#endif
#endif

typedef __attribute__((ext_vector_type(16))) _Float16 v16h;
typedef __attribute__((ext_vector_type(8)))  _Float16 v8h;
typedef __attribute__((ext_vector_type(4)))  _Float16 v4h;
typedef __attribute__((ext_vector_type(8)))  float    v8f;
typedef __attribute__((ext_vector_type(4)))  float    v4f;
typedef __attribute__((ext_vector_type(4)))  int      v4i;
typedef __attribute__((address_space(3)))    v4i      as3_v4i;

// --------------------------- f32 -> f16 convert ----------------------------
__global__ void cvt_f32_f16_x4(const v4f* __restrict__ in,
                               v4h* __restrict__ out, size_t n4) {
    size_t i = (size_t)blockIdx.x * blockDim.x + threadIdx.x;
    if (i < n4) out[i] = __builtin_convertvector(in[i], v4h);
}

// ------------------------------ fc1 via WMMA -------------------------------
// Workgroup: 8 waves, tile M=256 x N=32, shared LDS x tile [32t][32k] stored
// TRANSPOSED (k contiguous). Each wave: 2x2 subtiles (32M x 32N), 4 WMMAs per
// K-step of 32. Split-K over blockIdx.z; partials to u1p[kc].
__global__ void __launch_bounds__(256)
fc1_wmma_kernel(const _Float16* __restrict__ Xh,   // [B][K_IN][T_LEN]
                const _Float16* __restrict__ Wh,   // [N_HID][K_IN]
                float* __restrict__ u1p) {         // [KSPLIT][B][N_HID][T_LEN]
    __shared__ _Float16 xsT[32 * XPITCH];          // [t][k], pitch 40 halves

    const int tid  = threadIdx.x;
    const int lane = tid & 31;
    const int wave = tid >> 5;                     // 0..7
    const int half = lane >> 4;
    const int row  = lane & 15;
    const int col  = lane & 15;

    const int t0 = blockIdx.x * 32;
    const int m0 = blockIdx.y * 256 + wave * 32;
    const int b  = blockIdx.z / KSPLIT;
    const int kc = blockIdx.z % KSPLIT;
    const int kbeg = kc * KCH;
    const int kend = kbeg + KCH;

    const _Float16* xb = Xh + (size_t)b * K_IN * T_LEN;

    // staging assignment: thread -> (t, 4 consecutive k)
    const int s_t  = tid & 31;
    const int s_k4 = (tid >> 5) * 4;               // 0,4,...,28
    int tg = t0 + s_t;
    tg = (tg < T_LEN) ? tg : (T_LEN - 1);          // clamp; OOB cols unstored
    const _Float16* xcol = xb + tg;

    v8f c00 = {}, c01 = {}, c10 = {}, c11 = {};

    for (int k0 = kbeg; k0 < kend; k0 += 32) {
        // ---- stage x tile transposed: LDS[t][k0+s_k4..+3] ----
        v4h st;
        st[0] = xcol[(size_t)(k0 + s_k4 + 0) * T_LEN];
        st[1] = xcol[(size_t)(k0 + s_k4 + 1) * T_LEN];
        st[2] = xcol[(size_t)(k0 + s_k4 + 2) * T_LEN];
        st[3] = xcol[(size_t)(k0 + s_k4 + 3) * T_LEN];
        *(v4h*)(&xsT[s_t * XPITCH + s_k4]) = st;
        __syncthreads();

        // ---- A fragments (16-bit A 16x32 layout): 2 m-subtiles ----
        const _Float16* wr0 = Wh + (size_t)(m0 + row)      * K_IN + k0 + half * 16;
        const _Float16* wr1 = Wh + (size_t)(m0 + 16 + row) * K_IN + k0 + half * 16;
        __builtin_prefetch((const void*)(wr0 + 256), 0, 3); // W stream, 8 steps ahead
        __builtin_prefetch((const void*)(wr1 + 256), 0, 3);
        v8h a0lo = *(const v8h*)(wr0);
        v8h a0hi = *(const v8h*)(wr0 + 8);
        v8h a1lo = *(const v8h*)(wr1);
        v8h a1hi = *(const v8h*)(wr1 + 8);
        v16h a0 = __builtin_shufflevector(a0lo, a0hi,
                    0,1,2,3,4,5,6,7,8,9,10,11,12,13,14,15);
        v16h a1 = __builtin_shufflevector(a1lo, a1hi,
                    0,1,2,3,4,5,6,7,8,9,10,11,12,13,14,15);

        // ---- B fragments: contiguous 16 halves per lane (2 x ds_load_b128) --
        const _Float16* bp0 = &xsT[(col)      * XPITCH + half * 16];
        const _Float16* bp1 = &xsT[(16 + col) * XPITCH + half * 16];
        v8h b0lo = *(const v8h*)(bp0);
        v8h b0hi = *(const v8h*)(bp0 + 8);
        v8h b1lo = *(const v8h*)(bp1);
        v8h b1hi = *(const v8h*)(bp1 + 8);
        v16h bf0 = __builtin_shufflevector(b0lo, b0hi,
                    0,1,2,3,4,5,6,7,8,9,10,11,12,13,14,15);
        v16h bf1 = __builtin_shufflevector(b1lo, b1hi,
                    0,1,2,3,4,5,6,7,8,9,10,11,12,13,14,15);

        c00 = __builtin_amdgcn_wmma_f32_16x16x32_f16(false, a0, false, bf0,
                                                     (short)0, c00, false, false);
        c01 = __builtin_amdgcn_wmma_f32_16x16x32_f16(false, a0, false, bf1,
                                                     (short)0, c01, false, false);
        c10 = __builtin_amdgcn_wmma_f32_16x16x32_f16(false, a1, false, bf0,
                                                     (short)0, c10, false, false);
        c11 = __builtin_amdgcn_wmma_f32_16x16x32_f16(false, a1, false, bf1,
                                                     (short)0, c11, false, false);
        __syncthreads();
    }

    // ---- store partials: VGPR r -> M = half*8 + r, N = col ----
    float* up = u1p + ((size_t)kc * B_SZ + b) * N_HID * T_LEN;
    const int t_a = t0 + col;
    const int t_b = t0 + 16 + col;
#pragma unroll
    for (int r = 0; r < 8; ++r) {
        const int ma = m0 + half * 8 + r;
        const int mb = ma + 16;
        if (t_a < T_LEN) {
            up[(size_t)ma * T_LEN + t_a] = c00[r];
            up[(size_t)mb * T_LEN + t_a] = c10[r];
        }
        if (t_b < T_LEN) {
            up[(size_t)ma * T_LEN + t_b] = c01[r];
            up[(size_t)mb * T_LEN + t_b] = c11[r];
        }
    }
}

// ----------------------- split-K partial reduction -------------------------
__global__ void reduce_k4(const float* __restrict__ up, float* __restrict__ u,
                          size_t n) {
    size_t i = (size_t)blockIdx.x * blockDim.x + threadIdx.x;
    if (i < n)
        u[i] = up[i] + up[n + i] + up[2 * n + i] + up[3 * n + i];
}

// ------------------------------ fc2 via WMMA -------------------------------
// Tiny GEMM: M=19 (2 clamped tiles), K=512, N=300 per batch. One wave/tile.
__global__ void __launch_bounds__(32)
fc2_wmma_kernel(const _Float16* __restrict__ S1h,  // [B][N_HID][T_LEN]
                const _Float16* __restrict__ W2h,  // [N_OUT][N_HID]
                float* __restrict__ u2) {          // [B][N_OUT][T_LEN]
    __shared__ _Float16 xs[32 * 16];
    const int lane = threadIdx.x;
    const int half = lane >> 4;
    const int row  = lane & 15;
    const int col  = lane & 15;
    const int t0 = blockIdx.x * 16;
    const int m0 = blockIdx.y * 16;
    const int b  = blockIdx.z;
    const _Float16* xb = S1h + (size_t)b * N_HID * T_LEN;

    int mr = m0 + row;                     // clamp OOB rows (M=19)
    mr = (mr < N_OUT) ? mr : (N_OUT - 1);

    v8f c = {};
    for (int k0 = 0; k0 < N_HID; k0 += 32) {
        const _Float16* xr = xb + (size_t)(k0 + lane) * T_LEN;
#pragma unroll
        for (int j = 0; j < 16; ++j) {
            int tj = t0 + j;
            tj = (tj < T_LEN) ? tj : (T_LEN - 1);
            xs[lane * 16 + j] = xr[tj];
        }
        __syncthreads();

        const _Float16* wr = W2h + (size_t)mr * N_HID + k0 + half * 16;
        v8h alo = *(const v8h*)(wr);
        v8h ahi = *(const v8h*)(wr + 8);
        v16h a = __builtin_shufflevector(alo, ahi,
                    0,1,2,3,4,5,6,7,8,9,10,11,12,13,14,15);
        v16h bf;
#pragma unroll
        for (int j = 0; j < 16; ++j)
            bf[j] = xs[(half * 16 + j) * 16 + col];

        c = __builtin_amdgcn_wmma_f32_16x16x32_f16(false, a, false, bf,
                                                   (short)0, c, false, false);
        __syncthreads();
    }

    const int t = t0 + col;
    if (t < T_LEN) {
#pragma unroll
        for (int r = 0; r < 8; ++r) {
            int m = m0 + half * 8 + r;
            if (m < N_OUT)
                u2[((size_t)b * N_OUT + m) * T_LEN + t] = c[r];
        }
    }
}

// --------------------- fused psp (FIR) + spike recurrence ------------------
// One thread per (b, neuron) row. 128-slot circular u-history in LDS, filled
// 4 samples at a time by async global->LDS b128 copies (ASYNCcnt), pipelined
// one group ahead; 64-tap FIR + event-driven refractory ring.
__global__ void __launch_bounds__(PSP_BLK)
psp_spike_kernel(const float* __restrict__ u, float* __restrict__ s,
                 int rows, int T) {
    __shared__ float eps_s[KTAPS];
    __shared__ float nu_s[KTAPS];
    __shared__ float hist[PSP_BLK * HPITCH];   // circular u history (128 slots)
    __shared__ float acc[PSP_BLK * APITCH];    // refractory ring (64 slots)

    const int tid = threadIdx.x;
    if (tid < KTAPS) {
        float tk = (float)tid;
        eps_s[tid] = (tk / TAU_SR) * __expf(1.0f - tk / TAU_SR);
        nu_s[tid]  = -2.0f * THETA * tk * __expf(1.0f - tk);
    }
#pragma unroll 8
    for (int k = 0; k < HSLOTS; ++k) hist[tid * HPITCH + k] = 0.0f;
#pragma unroll 8
    for (int k = 0; k < KTAPS; ++k)  acc[tid * APITCH + k] = 0.0f;
    __syncthreads();

    const int r = blockIdx.x * blockDim.x + tid;
    if (r >= rows) return;

    const float* ur = u + (size_t)r * T;
    float* sr       = s + (size_t)r * T;
    float* hrow     = &hist[tid * HPITCH];
    float* arow     = &acc[tid * APITCH];

#if HAVE_ASYNC_LDS
    // prologue: async-copy u[0..3] into history slots 0..3 (per-lane LDS dest)
    __builtin_amdgcn_global_load_async_to_lds_b128(
        (v4i*)(ur), (as3_v4i*)(void*)(hrow), 0, 0);
#endif

    int base = 0;
    for (int t = 0; t < T; ++t) {
        if ((t & 3) == 0) {
#if HAVE_ASYNC_LDS
            __builtin_amdgcn_s_wait_asynccnt(0);   // group [t..t+3] landed
            if (t + 4 < T)                          // prefetch next group
                __builtin_amdgcn_global_load_async_to_lds_b128(
                    (v4i*)(ur + t + 4),
                    (as3_v4i*)(void*)(hrow + ((t + 4) & (HSLOTS - 1))),
                    0, 0);
#else
#pragma unroll
            for (int d = 0; d < 4; ++d)
                hrow[(t + d) & (HSLOTS - 1)] = ur[t + d];
#endif
        }

        float y = 0.0f;
#pragma unroll 8
        for (int k = 0; k < KTAPS; ++k)
            y += eps_s[k] * hrow[(t - k) & (HSLOTS - 1)];

        float v  = y + arow[base];
        float sp = (v >= THETA) ? 1.0f : 0.0f;
        if (sp != 0.0f) {
#pragma unroll 8
            for (int k = 1; k < KTAPS; ++k)
                arow[(base + k) & 63] += nu_s[k];
        }
        arow[base] = 0.0f;                 // retire slot leaving the window
        base = (base + 1) & 63;
        sr[t] = sp;
    }
}

// ---------------------------------------------------------------------------
extern "C" void kernel_launch(void* const* d_in, const int* in_sizes, int n_in,
                              void* d_out, int out_size, void* d_ws, size_t ws_size,
                              hipStream_t stream) {
    const float* x  = (const float*)d_in[0];   // [8,2,128,128,300]
    const float* W1 = (const float*)d_in[1];   // [512, 32768]
    const float* W2 = (const float*)d_in[2];   // [19, 512]
    float* out = (float*)d_out;                // [8, 19, 300]

    const size_t NX  = (size_t)B_SZ * K_IN * T_LEN;    // 78,643,200
    const size_t NW1 = (size_t)N_HID * K_IN;           // 16,777,216
    const size_t NW2 = (size_t)N_OUT * N_HID;          // 9,728
    const size_t NU1 = (size_t)B_SZ * N_HID * T_LEN;   // 1,228,800
    const size_t NU2 = (size_t)B_SZ * N_OUT * T_LEN;   // 45,600

    char* ws = (char*)d_ws;
    size_t off = 0;
    auto carve = [&](size_t bytes) -> void* {
        void* p = ws + off;
        off = (off + bytes + 255) & ~(size_t)255;
        return p;
    };
    _Float16* xh  = (_Float16*)carve(NX  * sizeof(_Float16));
    _Float16* w1h = (_Float16*)carve(NW1 * sizeof(_Float16));
    _Float16* w2h = (_Float16*)carve(NW2 * sizeof(_Float16));
    float*    u1p = (float*)carve(KSPLIT * NU1 * sizeof(float));
    float*    u1  = (float*)carve(NU1 * sizeof(float));
    float*    s1  = (float*)carve(NU1 * sizeof(float));
    _Float16* s1h = (_Float16*)carve(NU1 * sizeof(_Float16));
    float*    u2  = (float*)carve(NU2 * sizeof(float));

    // 1) downconvert x, W1, W2 to f16 (spikes exact in f16)
    cvt_f32_f16_x4<<<(unsigned)((NX / 4 + 255) / 256), 256, 0, stream>>>(
        (const v4f*)x, (v4h*)xh, NX / 4);
    cvt_f32_f16_x4<<<(unsigned)((NW1 / 4 + 255) / 256), 256, 0, stream>>>(
        (const v4f*)W1, (v4h*)w1h, NW1 / 4);
    cvt_f32_f16_x4<<<(unsigned)((NW2 / 4 + 255) / 256), 256, 0, stream>>>(
        (const v4f*)W2, (v4h*)w2h, NW2 / 4);

    // 2) fc1 WMMA: grid = (ceil(300/32)=10 n-blocks, 2 m-blocks, B*KSPLIT)
    fc1_wmma_kernel<<<dim3(10, 2, B_SZ * KSPLIT), 256, 0, stream>>>(
        xh, w1h, u1p);
    reduce_k4<<<(unsigned)((NU1 + 255) / 256), 256, 0, stream>>>(u1p, u1, NU1);

    // 3) psp + spike, layer 1
    psp_spike_kernel<<<(B_SZ * N_HID) / PSP_BLK, PSP_BLK, 0, stream>>>(
        u1, s1, B_SZ * N_HID, T_LEN);

    // 4) fc2 WMMA (s1 -> f16 first)
    cvt_f32_f16_x4<<<(unsigned)((NU1 / 4 + 255) / 256), 256, 0, stream>>>(
        (const v4f*)s1, (v4h*)s1h, NU1 / 4);
    fc2_wmma_kernel<<<dim3((T_LEN + 15) / 16, 2, B_SZ), 32, 0, stream>>>(
        s1h, w2h, u2);

    // 5) psp + spike, layer 2 -> output
    psp_spike_kernel<<<(B_SZ * N_OUT + PSP_BLK - 1) / PSP_BLK, PSP_BLK, 0, stream>>>(
        u2, out, B_SZ * N_OUT, T_LEN);
}
